// QFormerImproved_54331336295041
// MI455X (gfx1250) — matrix operand order repaired and checked by reading, not compiled, not run on hardware
//
#include <hip/hip_runtime.h>
#include <math.h>

#define D      1024
#define HEADS  16
#define DH     64
#define BATCH  256
#define SQ     64
#define NOBJ   100
#define NRELS  200
#define HOPS   4

typedef unsigned short u16;
typedef __attribute__((ext_vector_type(16))) __bf16 v16bf;
typedef __attribute__((ext_vector_type(8)))  float  v8f;

__device__ __forceinline__ float bf2f(u16 x) {
    unsigned int u = ((unsigned int)x) << 16;
    float f; __builtin_memcpy(&f, &u, 4); return f;
}
__device__ __forceinline__ u16 f2bf(float f) {
    unsigned int u; __builtin_memcpy(&u, &f, 4);
    u += 0x7fffu + ((u >> 16) & 1u);   // round-to-nearest-even
    return (u16)(u >> 16);
}

// ---------------------------------------------------------------- elementwise
__global__ void cvt_kernel(u16* __restrict__ dst, const float* __restrict__ src, size_t n) {
    size_t i = (size_t)blockIdx.x * blockDim.x + threadIdx.x;
    size_t st = (size_t)gridDim.x * blockDim.x;
    for (; i < n; i += st) dst[i] = f2bf(src[i]);
}

__global__ void zero_kernel(float* __restrict__ p, size_t n) {
    size_t i = (size_t)blockIdx.x * blockDim.x + threadIdx.x;
    size_t st = (size_t)gridDim.x * blockDim.x;
    for (; i < n; i += st) p[i] = 0.f;
}

__global__ void bcast_kernel(float* __restrict__ out, u16* __restrict__ outbf,
                             const float* __restrict__ row, size_t n) {
    size_t i = (size_t)blockIdx.x * blockDim.x + threadIdx.x;
    size_t st = (size_t)gridDim.x * blockDim.x;
    for (; i < n; i += st) {
        float v = row[i % D];
        out[i] = v;
        if (outbf) outbf[i] = f2bf(v);
    }
}

// out = x + rowv[col]   (c_in = control + hop_emb[hop])
__global__ void addrow_kernel(float* __restrict__ out, u16* __restrict__ outbf,
                              const float* __restrict__ x, const float* __restrict__ rowv,
                              size_t n) {
    size_t i = (size_t)blockIdx.x * blockDim.x + threadIdx.x;
    size_t st = (size_t)gridDim.x * blockDim.x;
    for (; i < n; i += st) {
        float v = x[i] + rowv[i % D];
        out[i] = v;
        if (outbf) outbf[i] = f2bf(v);
    }
}

// dst row = [a(D) | b(D)]
__global__ void pack2_kernel(u16* __restrict__ dst, const u16* __restrict__ a,
                             const u16* __restrict__ b, size_t rows) {
    size_t n = rows * (size_t)(2 * D);
    size_t i = (size_t)blockIdx.x * blockDim.x + threadIdx.x;
    size_t st = (size_t)gridDim.x * blockDim.x;
    for (; i < n; i += st) {
        size_t r = i / (2 * D); int c = (int)(i % (2 * D));
        dst[i] = (c < D) ? a[r * D + c] : b[r * D + (c - D)];
    }
}

// dst row = [a(D) | b(D) | c(D)]
__global__ void pack3_kernel(u16* __restrict__ dst, const u16* __restrict__ a,
                             const u16* __restrict__ b, const u16* __restrict__ c3,
                             size_t rows) {
    size_t n = rows * (size_t)(3 * D);
    size_t i = (size_t)blockIdx.x * blockDim.x + threadIdx.x;
    size_t st = (size_t)gridDim.x * blockDim.x;
    for (; i < n; i += st) {
        size_t r = i / (3 * D); int c = (int)(i % (3 * D));
        dst[i] = (c < D) ? a[r * D + c] : (c < 2 * D ? b[r * D + (c - D)] : c3[r * D + (c - 2 * D)]);
    }
}

// ---------------------------------------------------------------- WMMA GEMM
// Y[R,C] = X[R,K] * W[C,K]^T + bias ; EPI: 0=none 1=gelu(exact) 2=sigmoid
// One wave per 32x64 tile: 2 A-frags x 4 B-frags -> 8 WMMAs per K-step.
// Software-pipelined: next K-step fragments are issued before current WMMAs,
// so the scheduler can wait on a partial loadcnt instead of draining to 0.
union FragU { uint4 u[2]; v16bf v; };

template <int EPI>
__global__ __launch_bounds__(32) void gemm_k(
    const u16* __restrict__ X, const u16* __restrict__ W,
    const float* __restrict__ bias, float* __restrict__ Y,
    u16* __restrict__ Ybf, int R, int C, int K) {
    const int lane = threadIdx.x;
    const int l15 = lane & 15;
    const int hi  = lane >> 4;                    // lane half
    const int tileR = blockIdx.y << 5;            // 32 rows
    const int tileC = blockIdx.x << 6;            // 64 cols
    v8f acc[2][4];
#pragma unroll
    for (int m = 0; m < 2; ++m)
#pragma unroll
        for (int j = 0; j < 4; ++j) acc[m][j] = v8f{};

    // A 16x32 frag: elem i -> K = kb + (i<8?0:16) + hi*8 + (i&7)
    const u16* xp0 = X + (size_t)(tileR + l15) * K + hi * 8;
    const u16* xp1 = xp0 + (size_t)16 * K;
    // B 32x16 frag: lane = column (W row), elem i -> K = kb + (i<8?0:16) + hi*8 + (i&7)
    const u16* wq0 = W + (size_t)(tileC + l15) * K + hi * 8;
    const u16* wq1 = wq0 + (size_t)16 * K;
    const u16* wq2 = wq0 + (size_t)32 * K;
    const u16* wq3 = wq0 + (size_t)48 * K;

    auto ldA = [&](const u16* p, int kb, FragU& f) {
        f.u[0] = *reinterpret_cast<const uint4*>(p + kb);
        f.u[1] = *reinterpret_cast<const uint4*>(p + kb + 16);
    };
    auto ldB = [&](const u16* p, int kb, FragU& f) {
        f.u[0] = *reinterpret_cast<const uint4*>(p + kb);
        f.u[1] = *reinterpret_cast<const uint4*>(p + kb + 16);
    };

    FragU a0, a1, b0, b1, b2, b3;
    ldA(xp0, 0, a0); ldA(xp1, 0, a1);
    ldB(wq0, 0, b0); ldB(wq1, 0, b1); ldB(wq2, 0, b2); ldB(wq3, 0, b3);

    int kb = 0;
    for (; kb + 32 < K; kb += 32) {
        FragU na0, na1, nb0, nb1, nb2, nb3;
        const int nk = kb + 32;
        ldA(xp0, nk, na0); ldA(xp1, nk, na1);
        ldB(wq0, nk, nb0); ldB(wq1, nk, nb1); ldB(wq2, nk, nb2); ldB(wq3, nk, nb3);
        acc[0][0] = __builtin_amdgcn_wmma_f32_16x16x32_bf16(false, a0.v, false, b0.v, (short)0, acc[0][0], false, false);
        acc[1][0] = __builtin_amdgcn_wmma_f32_16x16x32_bf16(false, a1.v, false, b0.v, (short)0, acc[1][0], false, false);
        acc[0][1] = __builtin_amdgcn_wmma_f32_16x16x32_bf16(false, a0.v, false, b1.v, (short)0, acc[0][1], false, false);
        acc[1][1] = __builtin_amdgcn_wmma_f32_16x16x32_bf16(false, a1.v, false, b1.v, (short)0, acc[1][1], false, false);
        acc[0][2] = __builtin_amdgcn_wmma_f32_16x16x32_bf16(false, a0.v, false, b2.v, (short)0, acc[0][2], false, false);
        acc[1][2] = __builtin_amdgcn_wmma_f32_16x16x32_bf16(false, a1.v, false, b2.v, (short)0, acc[1][2], false, false);
        acc[0][3] = __builtin_amdgcn_wmma_f32_16x16x32_bf16(false, a0.v, false, b3.v, (short)0, acc[0][3], false, false);
        acc[1][3] = __builtin_amdgcn_wmma_f32_16x16x32_bf16(false, a1.v, false, b3.v, (short)0, acc[1][3], false, false);
        a0 = na0; a1 = na1; b0 = nb0; b1 = nb1; b2 = nb2; b3 = nb3;
    }
    // final K-step (no prefetch)
    acc[0][0] = __builtin_amdgcn_wmma_f32_16x16x32_bf16(false, a0.v, false, b0.v, (short)0, acc[0][0], false, false);
    acc[1][0] = __builtin_amdgcn_wmma_f32_16x16x32_bf16(false, a1.v, false, b0.v, (short)0, acc[1][0], false, false);
    acc[0][1] = __builtin_amdgcn_wmma_f32_16x16x32_bf16(false, a0.v, false, b1.v, (short)0, acc[0][1], false, false);
    acc[1][1] = __builtin_amdgcn_wmma_f32_16x16x32_bf16(false, a1.v, false, b1.v, (short)0, acc[1][1], false, false);
    acc[0][2] = __builtin_amdgcn_wmma_f32_16x16x32_bf16(false, a0.v, false, b2.v, (short)0, acc[0][2], false, false);
    acc[1][2] = __builtin_amdgcn_wmma_f32_16x16x32_bf16(false, a1.v, false, b2.v, (short)0, acc[1][2], false, false);
    acc[0][3] = __builtin_amdgcn_wmma_f32_16x16x32_bf16(false, a0.v, false, b3.v, (short)0, acc[0][3], false, false);
    acc[1][3] = __builtin_amdgcn_wmma_f32_16x16x32_bf16(false, a1.v, false, b3.v, (short)0, acc[1][3], false, false);

#pragma unroll
    for (int m = 0; m < 2; ++m) {
#pragma unroll
        for (int j = 0; j < 4; ++j) {
            int col = tileC + j * 16 + l15;
            float bv = bias ? bias[col] : 0.f;
#pragma unroll
            for (int r = 0; r < 8; ++r) {
                int row = tileR + m * 16 + r + hi * 8;   // C/D layout: VGPR r, lane half
                float v = acc[m][j][r] + bv;
                if (EPI == 1) v = 0.5f * v * (1.f + erff(v * 0.70710678118654752f));
                else if (EPI == 2) v = 1.f / (1.f + expf(-v));
                size_t o = (size_t)row * C + col;
                if (Y)   Y[o] = v;
                if (Ybf) Ybf[o] = f2bf(v);
            }
        }
    }
}

// ---------------------------------------------------------------- attention
// One block per (b,h), 64 threads. Q f32 (B,D); K/V bf16 (B,Skv,D).
// ctx: bf16 (B,D). avgw: head-averaged weights accumulated via atomics.
__global__ __launch_bounds__(64) void attn_kernel(
    const float* __restrict__ Q, const u16* __restrict__ Kp,
    const u16* __restrict__ Vp, const int* __restrict__ mask,
    u16* __restrict__ ctx, float* __restrict__ avgw, int Skv) {
    __shared__ float qs[DH];
    __shared__ float sc[256];
    int b = blockIdx.x >> 4;
    int h = blockIdx.x & 15;
    int t = threadIdx.x;
    qs[t] = Q[(size_t)b * D + h * DH + t];
    __syncthreads();
    for (int s = t; s < Skv; s += 64) {
        const u16* kp = Kp + ((size_t)b * Skv + s) * D + h * DH;
        float acc = 0.f;
        for (int i = 0; i < DH; ++i) acc += qs[i] * bf2f(kp[i]);
        acc *= 0.125f;                               // 1/sqrt(64)
        if (mask && mask[b * SQ + s] == 0) acc = -INFINITY;
        sc[s] = acc;
    }
    __syncthreads();
    float m = -INFINITY;
    for (int s = 0; s < Skv; ++s) m = fmaxf(m, sc[s]);
    __syncthreads();
    for (int s = t; s < Skv; s += 64) sc[s] = expf(sc[s] - m);
    __syncthreads();
    float sum = 0.f;
    for (int s = 0; s < Skv; ++s) sum += sc[s];
    float inv = 1.f / sum;
    float o = 0.f;
    const u16* vp = Vp + (size_t)b * Skv * D + h * DH + t;
    for (int s = 0; s < Skv; ++s) o += sc[s] * bf2f(vp[(size_t)s * D]);
    ctx[(size_t)b * D + h * DH + t] = f2bf(o * inv);
    if (avgw) {
        for (int s = t; s < Skv; s += 64)
            atomicAdd(&avgw[(size_t)b * Skv + s], sc[s] * inv * (1.0f / HEADS));
    }
}

// ---------------------------------------------------------------- layernorm
// x = gate ? gate*A + (1-gate)*B : (B ? A+B : A);  y = LN(x)*gamma + beta
__global__ __launch_bounds__(256) void ln_kernel(
    const float* __restrict__ A, const float* __restrict__ Bp,
    const float* __restrict__ gate, const float* __restrict__ gamma,
    const float* __restrict__ beta, float* __restrict__ out,
    float* __restrict__ out2, u16* __restrict__ outbf) {
    __shared__ float ssum[256], ssq[256];
    int row = blockIdx.x, t = threadIdx.x;
    size_t base = (size_t)row * D;
    float x[4];
#pragma unroll
    for (int i = 0; i < 4; ++i) {
        int c = t + i * 256;
        float a = A[base + c], v;
        if (gate) { float g = gate[base + c]; v = g * a + (1.f - g) * Bp[base + c]; }
        else if (Bp) v = a + Bp[base + c];
        else v = a;
        x[i] = v;
    }
    float ls = 0.f, lq = 0.f;
#pragma unroll
    for (int i = 0; i < 4; ++i) { ls += x[i]; lq += x[i] * x[i]; }
    ssum[t] = ls; ssq[t] = lq;
    __syncthreads();
    for (int st = 128; st > 0; st >>= 1) {
        if (t < st) { ssum[t] += ssum[t + st]; ssq[t] += ssq[t + st]; }
        __syncthreads();
    }
    float mean = ssum[0] * (1.f / D);
    float var  = ssq[0] * (1.f / D) - mean * mean;   // biased, matches torch LN
    float rstd = rsqrtf(var + 1e-5f);
#pragma unroll
    for (int i = 0; i < 4; ++i) {
        int c = t + i * 256;
        float y = (x[i] - mean) * rstd * gamma[c] + beta[c];
        out[base + c] = y;
        if (out2)  out2[base + c] = y;
        if (outbf) outbf[base + c] = f2bf(y);
    }
}

// ---------------------------------------------------------------- host
extern "C" void kernel_launch(void* const* d_in, const int* in_sizes, int n_in,
                              void* d_out, int out_size, void* d_ws, size_t ws_size,
                              hipStream_t stream) {
    (void)in_sizes; (void)n_in; (void)out_size; (void)ws_size;
    // --- inputs
    const float* qe    = (const float*)d_in[0];
    const float* obj   = (const float*)d_in[1];
    const float* rel   = (const float*)d_in[2];
    const int*   qmask = (const int*)d_in[3];
    // --- params (jax pytree flatten order: nested dicts, keys sorted)
    const float* ca_b_in  = (const float*)d_in[4];   // ctrl.attn.b_in (3D)
    const float* ca_b_out = (const float*)d_in[5];
    const float* ca_w_in  = (const float*)d_in[6];   // (3D,D)
    const float* ca_w_out = (const float*)d_in[7];
    const float* c_ln_b   = (const float*)d_in[8];
    const float* c_ln_g   = (const float*)d_in[9];
    const float* c_proj_b = (const float*)d_in[10];
    const float* c_proj_w = (const float*)d_in[11];
    const float* c_u1_b   = (const float*)d_in[12];
    const float* c_u1_w   = (const float*)d_in[13];  // (D,2D)
    const float* c_u2_b   = (const float*)d_in[14];
    const float* c_u2_w   = (const float*)d_in[15];
    const float* hop_emb  = (const float*)d_in[16];
    const float* init_c   = (const float*)d_in[17];
    const float* init_m   = (const float*)d_in[18];
    const float* f_ln_b   = (const float*)d_in[19];
    const float* f_ln_g   = (const float*)d_in[20];
    const float* o1_b = (const float*)d_in[21];
    const float* o1_w = (const float*)d_in[22];      // (D,2D)
    const float* o2_b = (const float*)d_in[23];
    const float* o2_w = (const float*)d_in[24];
    const float* rc1_b = (const float*)d_in[25];
    const float* rc1_w = (const float*)d_in[26];     // (D,2D)
    const float* rc2_b = (const float*)d_in[27];
    const float* rc2_w = (const float*)d_in[28];
    const float* rg_b  = (const float*)d_in[29];
    const float* rg_w  = (const float*)d_in[30];     // (D,2D)
    const float* r_ln_b = (const float*)d_in[31];
    const float* r_ln_g = (const float*)d_in[32];
    const float* oa_b_in  = (const float*)d_in[33];
    const float* oa_b_out = (const float*)d_in[34];
    const float* oa_w_in  = (const float*)d_in[35];
    const float* oa_w_out = (const float*)d_in[36];
    const float* ra_b_in  = (const float*)d_in[37];
    const float* ra_b_out = (const float*)d_in[38];
    const float* ra_w_in  = (const float*)d_in[39];
    const float* ra_w_out = (const float*)d_in[40];
    const float* wg_b = (const float*)d_in[41];
    const float* wg_w = (const float*)d_in[42];      // (D,3D)
    const float* w_ln_b = (const float*)d_in[43];
    const float* w_ln_g = (const float*)d_in[44];
    const float* wt1_b = (const float*)d_in[45];
    const float* wt1_w = (const float*)d_in[46];     // (D,2D)
    const float* wt2_b = (const float*)d_in[47];
    const float* wt2_w = (const float*)d_in[48];

    const size_t DD  = (size_t)D * D;
    const size_t nQE = (size_t)BATCH * SQ * D;
    const size_t nOB = (size_t)BATCH * NOBJ * D;
    const size_t nRE = (size_t)BATCH * NRELS * D;
    const size_t nBD = (size_t)BATCH * D;

    // --- workspace carving
    char* wsb = (char*)d_ws;
    size_t off = 0;
    auto alloc = [&](size_t bytes) -> void* {
        off = (off + 255) & ~(size_t)255;
        void* p = wsb + off; off += bytes; return p;
    };
    u16* qe_bf  = (u16*)alloc(nQE * 2);
    u16* obj_bf = (u16*)alloc(nOB * 2);
    u16* rel_bf = (u16*)alloc(nRE * 2);
    u16* Kq = (u16*)alloc(nQE * 2); u16* Vq = (u16*)alloc(nQE * 2);
    u16* Ko = (u16*)alloc(nOB * 2); u16* Vo = (u16*)alloc(nOB * 2);
    u16* Kr = (u16*)alloc(nRE * 2); u16* Vr = (u16*)alloc(nRE * 2);
    u16* caw_in = (u16*)alloc(3 * DD * 2); u16* caw_out = (u16*)alloc(DD * 2);
    u16* oaw_in = (u16*)alloc(3 * DD * 2); u16* oaw_out = (u16*)alloc(DD * 2);
    u16* raw_in = (u16*)alloc(3 * DD * 2); u16* raw_out = (u16*)alloc(DD * 2);
    u16* projw = (u16*)alloc(DD * 2);
    u16* u1w = (u16*)alloc(2 * DD * 2); u16* u2w = (u16*)alloc(DD * 2);
    u16* rc1w = (u16*)alloc(2 * DD * 2); u16* rc2w = (u16*)alloc(DD * 2);
    u16* rgw = (u16*)alloc(2 * DD * 2);
    u16* wgw = (u16*)alloc(3 * DD * 2);
    u16* wt1w = (u16*)alloc(2 * DD * 2); u16* wt2w = (u16*)alloc(DD * 2);
    u16* o1wb = (u16*)alloc(2 * DD * 2); u16* o2wb = (u16*)alloc(DD * 2);
    u16* cinbf = (u16*)alloc(nBD * 2); u16* attbf = (u16*)alloc(nBD * 2);
    u16* ctxbf = (u16*)alloc(nBD * 2); u16* ctlbf = (u16*)alloc(nBD * 2);
    u16* oibf = (u16*)alloc(nBD * 2);  u16* ribf = (u16*)alloc(nBD * 2);
    u16* combbf = (u16*)alloc(nBD * 2); u16* membf = (u16*)alloc(nBD * 2);
    u16* robf = (u16*)alloc(nBD * 2);  u16* hidbf = (u16*)alloc(nBD * 2);
    u16* cat2 = (u16*)alloc(2 * nBD * 2); u16* cat3 = (u16*)alloc(3 * nBD * 2);
    float* Qtmp = (float*)alloc(nBD * 4); float* att = (float*)alloc(nBD * 4);
    float* cinf = (float*)alloc(nBD * 4); float* cmid = (float*)alloc(nBD * 4);
    float* paf = (float*)alloc(nBD * 4);  float* comb = (float*)alloc(nBD * 4);
    float* gatef = (float*)alloc(nBD * 4); float* updf = (float*)alloc(nBD * 4);
    float* rof = (float*)alloc(nBD * 4);  float* ctrlf = (float*)alloc(nBD * 4);
    float* memf = (float*)alloc(nBD * 4); float* finf = (float*)alloc(nBD * 4);

    // --- output layout: final | mems(4) | oas(4) | ras(4)
    float* outF = (float*)d_out;
    float* out_final = outF;
    float* out_mems = outF + nBD;
    float* out_oas = outF + nBD + 4 * nBD;
    float* out_ras = out_oas + (size_t)4 * BATCH * NOBJ;

    auto cvt = [&](u16* dst, const float* src, size_t n) {
        size_t bl = (n + 255) / 256; if (bl > 8192) bl = 8192;
        cvt_kernel<<<dim3((unsigned)bl), dim3(256), 0, stream>>>(dst, src, n);
    };
    auto gemm = [&](const u16* X, const u16* W, const float* bias, float* Y, u16* Ybf,
                    int R, int C, int K, int epi) {
        dim3 g(C / 64, R / 32), bl(32);
        if (epi == 1)      gemm_k<1><<<g, bl, 0, stream>>>(X, W, bias, Y, Ybf, R, C, K);
        else if (epi == 2) gemm_k<2><<<g, bl, 0, stream>>>(X, W, bias, Y, Ybf, R, C, K);
        else               gemm_k<0><<<g, bl, 0, stream>>>(X, W, bias, Y, Ybf, R, C, K);
    };

    // --- convert inputs & weights to bf16 (every call: deterministic)
    cvt(qe_bf, qe, nQE); cvt(obj_bf, obj, nOB); cvt(rel_bf, rel, nRE);
    cvt(caw_in, ca_w_in, 3 * DD); cvt(caw_out, ca_w_out, DD);
    cvt(oaw_in, oa_w_in, 3 * DD); cvt(oaw_out, oa_w_out, DD);
    cvt(raw_in, ra_w_in, 3 * DD); cvt(raw_out, ra_w_out, DD);
    cvt(projw, c_proj_w, DD);
    cvt(u1w, c_u1_w, 2 * DD); cvt(u2w, c_u2_w, DD);
    cvt(rc1w, rc1_w, 2 * DD); cvt(rc2w, rc2_w, DD);
    cvt(rgw, rg_w, 2 * DD);
    cvt(wgw, wg_w, 3 * DD);
    cvt(wt1w, wt1_w, 2 * DD); cvt(wt2w, wt2_w, DD);
    cvt(o1wb, o1_w, 2 * DD); cvt(o2wb, o2_w, DD);

    // --- hop-invariant K/V projections (packed in-proj slices: [wq;wk;wv])
    gemm(qe_bf,  caw_in + DD,     ca_b_in + D,     nullptr, Kq, BATCH * SQ,    D, D, 0);
    gemm(qe_bf,  caw_in + 2 * DD, ca_b_in + 2 * D, nullptr, Vq, BATCH * SQ,    D, D, 0);
    gemm(obj_bf, oaw_in + DD,     oa_b_in + D,     nullptr, Ko, BATCH * NOBJ,  D, D, 0);
    gemm(obj_bf, oaw_in + 2 * DD, oa_b_in + 2 * D, nullptr, Vo, BATCH * NOBJ,  D, D, 0);
    gemm(rel_bf, raw_in + DD,     ra_b_in + D,     nullptr, Kr, BATCH * NRELS, D, D, 0);
    gemm(rel_bf, raw_in + 2 * DD, ra_b_in + 2 * D, nullptr, Vr, BATCH * NRELS, D, D, 0);

    // --- init state + zero attn-weight accumulators in d_out
    bcast_kernel<<<1024, 256, 0, stream>>>(ctrlf, nullptr, init_c, nBD);
    bcast_kernel<<<1024, 256, 0, stream>>>(memf, membf, init_m, nBD);
    zero_kernel<<<512, 256, 0, stream>>>(out_oas, (size_t)4 * BATCH * (NOBJ + NRELS));

    for (int hop = 0; hop < HOPS; ++hop) {
        // control unit
        addrow_kernel<<<1024, 256, 0, stream>>>(cinf, cinbf, ctrlf, hop_emb + hop * D, nBD);
        gemm(cinbf, caw_in, ca_b_in, Qtmp, nullptr, BATCH, D, D, 0);              // q proj
        attn_kernel<<<BATCH * HEADS, 64, 0, stream>>>(Qtmp, Kq, Vq, qmask, ctxbf, nullptr, SQ);
        gemm(ctxbf, caw_out, ca_b_out, att, attbf, BATCH, D, D, 0);               // out proj
        pack2_kernel<<<1024, 256, 0, stream>>>(cat2, cinbf, attbf, BATCH);
        gemm(cat2, u1w, c_u1_b, nullptr, hidbf, BATCH, D, 2 * D, 1);              // gelu
        gemm(hidbf, u2w, c_u2_b, cmid, nullptr, BATCH, D, D, 0);
        gemm(attbf, projw, c_proj_b, paf, nullptr, BATCH, D, D, 0);
        ln_kernel<<<BATCH, 256, 0, stream>>>(cmid, paf, nullptr, c_ln_g, c_ln_b,
                                             ctrlf, nullptr, ctlbf);
        // read unit
        gemm(ctlbf, oaw_in, oa_b_in, Qtmp, nullptr, BATCH, D, D, 0);
        attn_kernel<<<BATCH * HEADS, 64, 0, stream>>>(Qtmp, Ko, Vo, nullptr, ctxbf,
                                                      out_oas + (size_t)hop * BATCH * NOBJ, NOBJ);
        gemm(ctxbf, oaw_out, oa_b_out, nullptr, oibf, BATCH, D, D, 0);
        gemm(ctlbf, raw_in, ra_b_in, Qtmp, nullptr, BATCH, D, D, 0);
        attn_kernel<<<BATCH * HEADS, 64, 0, stream>>>(Qtmp, Kr, Vr, nullptr, ctxbf,
                                                      out_ras + (size_t)hop * BATCH * NRELS, NRELS);
        gemm(ctxbf, raw_out, ra_b_out, nullptr, ribf, BATCH, D, D, 0);
        pack2_kernel<<<1024, 256, 0, stream>>>(cat2, oibf, ribf, BATCH);
        gemm(cat2, rc1w, rc1_b, nullptr, hidbf, BATCH, D, 2 * D, 1);
        gemm(hidbf, rc2w, rc2_b, comb, combbf, BATCH, D, D, 0);
        pack2_kernel<<<1024, 256, 0, stream>>>(cat2, combbf, membf, BATCH);
        gemm(cat2, rgw, rg_b, gatef, nullptr, BATCH, D, 2 * D, 2);                // sigmoid
        ln_kernel<<<BATCH, 256, 0, stream>>>(comb, memf, gatef, r_ln_g, r_ln_b,
                                             rof, nullptr, robf);
        // write unit
        pack3_kernel<<<1024, 256, 0, stream>>>(cat3, membf, robf, ctlbf, BATCH);
        gemm(cat3, wgw, wg_b, gatef, nullptr, BATCH, D, 3 * D, 2);                // sigmoid
        pack2_kernel<<<1024, 256, 0, stream>>>(cat2, robf, ctlbf, BATCH);
        gemm(cat2, wt1w, wt1_b, nullptr, hidbf, BATCH, D, 2 * D, 1);
        gemm(hidbf, wt2w, wt2_b, updf, nullptr, BATCH, D, D, 0);
        ln_kernel<<<BATCH, 256, 0, stream>>>(updf, memf, gatef, w_ln_g, w_ln_b,
                                             memf, out_mems + (size_t)hop * nBD, membf);
    }
    // final head
    pack2_kernel<<<1024, 256, 0, stream>>>(cat2, ctlbf, membf, BATCH);
    gemm(cat2, o1wb, o1_b, nullptr, hidbf, BATCH, D, 2 * D, 1);
    gemm(hidbf, o2wb, o2_b, finf, nullptr, BATCH, D, D, 0);
    ln_kernel<<<BATCH, 256, 0, stream>>>(finf, nullptr, nullptr, f_ln_g, f_ln_b,
                                         out_final, nullptr, nullptr);
}
